// VSSDecoderLayer_21182778704308
// MI455X (gfx1250) — matrix operand order repaired and checked by reading, not compiled, not run on hardware
//
#include <hip/hip_runtime.h>

// ---------------------------------------------------------------------------
// VSS decoder layer for gfx1250 (MI455X).
//  - GEMMs: register-blocked v_wmma_f32_16x16x32_bf16, one wave = 16x64 strip,
//    4 WMMAs per K-step, double-buffered fragment loads (loads for step s+1
//    issued before the WMMAs of step s) so s_wait_loadcnt overlaps with math.
//  - Selective scan: whole 48x1024 dts/B/C slice for a (b,k) staged into LDS
//    (192KB of the WGP's 320KB) via async global->LDS loads (ASYNCcnt), then a
//    barrier-free 1024-step recurrence on VALU (v_exp_f32).
// ---------------------------------------------------------------------------

typedef __attribute__((ext_vector_type(16))) __bf16 bf16x16;
typedef __attribute__((ext_vector_type(8)))  __bf16 bf16x8;
typedef __attribute__((ext_vector_type(8)))  float  f32x8;

#define DEV __device__ __forceinline__

// ---- problem dims ----
constexpr int Cc   = 256;
constexpr int Din  = 512;
constexpr int Nst  = 16;
constexpr int K4   = 4;
constexpr int Bb   = 4;
constexpr int Hh   = 32;
constexpr int Ww   = 32;
constexpr int Ll   = Hh * Ww;     // 1024
constexpr int Mr   = Bb * Ll;     // 4096 rows for all GEMMs
constexpr int Cdbl = 48;          // DT_RANK + 2*N_STATE
constexpr float EPSv = 1e-5f;

DEV __bf16 f2bf(float f) {
  unsigned u = __float_as_uint(f);
  u += 0x7fffu + ((u >> 16) & 1u);          // round-to-nearest-even
  unsigned short s = (unsigned short)(u >> 16);
  __bf16 r;
  __builtin_memcpy(&r, &s, 2);
  return r;
}
DEV float sigm(float x) { return 1.0f / (1.0f + __expf(-x)); }
DEV int   trmap(int l)  { return ((l & 31) << 5) | (l >> 5); }   // HxW transpose, H=W=32

// ---------------------------------------------------------------------------
// fp32 -> bf16 weight cast
// ---------------------------------------------------------------------------
__global__ __launch_bounds__(256)
void cast_bf16_kernel(const float* __restrict__ src, __bf16* __restrict__ dst, int n) {
  int i = blockIdx.x * 256 + threadIdx.x;
  if (i < n) dst[i] = f2bf(src[i]);
}

// ---------------------------------------------------------------------------
// LayerNorm over 256 cols -> bf16 rows (GEMM A operand)
// ---------------------------------------------------------------------------
__global__ __launch_bounds__(256)
void ln256_to_bf16_kernel(const float* __restrict__ x, const float* __restrict__ g,
                          const float* __restrict__ b, __bf16* __restrict__ out) {
  __shared__ float red[256];
  const int row = blockIdx.x, t = threadIdx.x;
  float v = x[(size_t)row * Cc + t];
  red[t] = v; __syncthreads();
  for (int s = 128; s > 0; s >>= 1) { if (t < s) red[t] += red[t + s]; __syncthreads(); }
  float mu = red[0] * (1.0f / Cc); __syncthreads();
  float d = v - mu;
  red[t] = d * d; __syncthreads();
  for (int s = 128; s > 0; s >>= 1) { if (t < s) red[t] += red[t + s]; __syncthreads(); }
  float rs = rsqrtf(red[0] * (1.0f / Cc) + EPSv);
  out[(size_t)row * Cc + t] = f2bf(d * rs * g[t] + b[t]);
}

// ---------------------------------------------------------------------------
// WMMA GEMM: C[M,N] = A[M,Kd](bf16, row-major) * W[N,Kd](bf16, row-major)^T
// Block = 256 threads = 8 waves. Each wave owns a 16(M) x 64(N) strip.
// Double-buffered: fragments for K-step s+1 are loaded before the 4 WMMAs of
// step s execute, hiding L2 latency behind matrix math.
// Fragment packing (this code's contiguous k order, same for A and B):
//   A: lane holds row (lane&15); halves 0..15 = K aklo..aklo+15,
//      aklo = lane<16 ? 0 : 16.
//   B: lane holds col (lane&15); halves 0..15 = K bklo..bklo+15,
//      bklo = lane<16 ? 0 : 16.  (W row-major in K -> cols of B are rows of W.)
// Requires: Kd multiple of 64, M of 128, N of 64.
// ---------------------------------------------------------------------------
union Frag { bf16x16 v; bf16x8 h[2]; };

__global__ __launch_bounds__(256)
void wmma_gemm_kernel(const __bf16* __restrict__ A, const __bf16* __restrict__ Wt,
                      float* __restrict__ Cout, int M, int N, int Kd) {
  const int lane = threadIdx.x & 31;
  const int wave = threadIdx.x >> 5;
  const int m0 = blockIdx.y * 128 + wave * 16;
  const int n0 = blockIdx.x * 64;
  const int r15 = lane & 15;
  const int aklo = (lane < 16) ? 0 : 16;
  const int bklo = (lane < 16) ? 0 : 16;

  const __bf16* arow = A + (size_t)(m0 + r15) * Kd;
  const __bf16* w0 = Wt + (size_t)(n0 + r15) * Kd + bklo;
  const __bf16* w1 = w0 + (size_t)16 * Kd;
  const __bf16* w2 = w0 + (size_t)32 * Kd;
  const __bf16* w3 = w0 + (size_t)48 * Kd;

  auto loadA = [&](Frag& f, int kb) {
    f.h[0] = *(const bf16x8*)(arow + kb + aklo);
    f.h[1] = *(const bf16x8*)(arow + kb + aklo + 8);
  };
  auto loadB4 = [&](Frag* f, int kb) {
    f[0].h[0] = *(const bf16x8*)(w0 + kb); f[0].h[1] = *(const bf16x8*)(w0 + kb + 8);
    f[1].h[0] = *(const bf16x8*)(w1 + kb); f[1].h[1] = *(const bf16x8*)(w1 + kb + 8);
    f[2].h[0] = *(const bf16x8*)(w2 + kb); f[2].h[1] = *(const bf16x8*)(w2 + kb + 8);
    f[3].h[0] = *(const bf16x8*)(w3 + kb); f[3].h[1] = *(const bf16x8*)(w3 + kb + 8);
  };

  f32x8 acc[4] = {{}, {}, {}, {}};
  Frag a0, a1, b0[4], b1[4];

  loadA(a0, 0);
  loadB4(b0, 0);

  // steady state: pairs of 32-deep K steps; Kd is a multiple of 64
  const int pairs = Kd / 64 - 1;
  for (int p = 0; p < pairs; ++p) {
    const int k1 = (2 * p + 1) * 32, k2 = (2 * p + 2) * 32;
    loadA(a1, k1);
    loadB4(b1, k1);
#pragma unroll
    for (int j = 0; j < 4; ++j)
      acc[j] = __builtin_amdgcn_wmma_f32_16x16x32_bf16(
          false, a0.v, false, b0[j].v, (short)0, acc[j], false, false);
    loadA(a0, k2);
    loadB4(b0, k2);
#pragma unroll
    for (int j = 0; j < 4; ++j)
      acc[j] = __builtin_amdgcn_wmma_f32_16x16x32_bf16(
          false, a1.v, false, b1[j].v, (short)0, acc[j], false, false);
  }
  // epilogue: last K step loads while the second-to-last computes
  loadA(a1, Kd - 32);
  loadB4(b1, Kd - 32);
#pragma unroll
  for (int j = 0; j < 4; ++j)
    acc[j] = __builtin_amdgcn_wmma_f32_16x16x32_bf16(
        false, a0.v, false, b0[j].v, (short)0, acc[j], false, false);
#pragma unroll
  for (int j = 0; j < 4; ++j)
    acc[j] = __builtin_amdgcn_wmma_f32_16x16x32_bf16(
        false, a1.v, false, b1[j].v, (short)0, acc[j], false, false);

  // C/D layout: VGPR v -> M = v (lanes 0-15) / v+8 (lanes 16-31); N = lane&15
  const int moff = (lane < 16) ? 0 : 8;
#pragma unroll
  for (int j = 0; j < 4; ++j)
#pragma unroll
    for (int v = 0; v < 8; ++v)
      Cout[(size_t)(m0 + moff + v) * N + n0 + j * 16 + r15] = acc[j][v];
}

// ---------------------------------------------------------------------------
// Depthwise 3x3 conv + bias + SiLU. In: xz rows (B*L,1024) cols 0..511 = xx.
// Out: xc[b][d][l] (l contiguous).
// ---------------------------------------------------------------------------
__global__ __launch_bounds__(256)
void conv_silu_kernel(const float* __restrict__ xz, const float* __restrict__ cw,
                      const float* __restrict__ cb, float* __restrict__ xc) {
  int idx = blockIdx.x * 256 + threadIdx.x;            // over B*Din*L
  int l = idx % Ll, d = (idx / Ll) % Din, b = idx / (Ll * Din);
  int h = l >> 5, w = l & 31;
  float acc = cb[d];
#pragma unroll
  for (int kh = 0; kh < 3; ++kh) {
#pragma unroll
    for (int kw = 0; kw < 3; ++kw) {
      int hh = h + kh - 1, ww = w + kw - 1;
      if (hh >= 0 && hh < Hh && ww >= 0 && ww < Ww) {
        acc += xz[((size_t)b * Ll + hh * Ww + ww) * (2 * Din) + d] * cw[d * 9 + kh * 3 + kw];
      }
    }
  }
  xc[idx] = acc * sigm(acc);
}

// ---------------------------------------------------------------------------
// x_dbl[b][k][c][l] = sum_d xs[b,k,d,l] * x_proj_w[k,c,d]  (c = 0..47)
// xs is an index-permutation of xc (never materialized).
// ---------------------------------------------------------------------------
__global__ __launch_bounds__(256)
void xproj_kernel(const float* __restrict__ xc, const float* __restrict__ xw,
                  float* __restrict__ xdbl) {
  int idx = blockIdx.x * 256 + threadIdx.x;            // over B*K*48*L
  int l = idx & (Ll - 1);
  int c = (idx >> 10) % Cdbl;
  int k = (idx / (Ll * Cdbl)) % K4;
  int b = idx / (Ll * Cdbl * K4);
  int li;
  switch (k) {
    case 0:  li = l; break;
    case 1:  li = trmap(l); break;
    case 2:  li = Ll - 1 - l; break;
    default: li = trmap(Ll - 1 - l); break;
  }
  const float* wrow = xw + ((size_t)k * Cdbl + c) * Din;
  const float* xcb  = xc + (size_t)b * Din * Ll;
  float acc = 0.f;
  for (int d = 0; d < Din; ++d) acc += xcb[(size_t)d * Ll + li] * wrow[d];
  xdbl[idx] = acc;
}

// ---------------------------------------------------------------------------
// Selective scan. One block = one (b,k): 512 threads, lane d = threadIdx.x.
// The whole 48x1024 float dts/B/C slice (192KB) is staged into LDS once via
// async global->LDS loads (ASYNCcnt), then the 1024-step recurrence runs with
// zero barriers: softplus delta, 16-state exp recurrence, y written with D*u.
// ---------------------------------------------------------------------------
__global__ __launch_bounds__(512)
void scan_kernel(const float* __restrict__ xc, const float* __restrict__ xdbl,
                 const float* __restrict__ dtw, const float* __restrict__ dtb,
                 const float* __restrict__ alogs, const float* __restrict__ dsc,
                 float* __restrict__ ys) {
  extern __shared__ float sxd[];                 // Cdbl*Ll floats = 192 KB
  const int k = blockIdx.x & 3, b = blockIdx.x >> 2;
  const int d = threadIdx.x;

  // ---- async bulk stage: global xdbl[b][k][:][:] -> LDS (b128 granules) ----
  const float* src = xdbl + ((size_t)(b * K4 + k) * Cdbl) * Ll;
  for (int i = threadIdx.x * 4; i < Cdbl * Ll; i += 512 * 4) {
    unsigned ldsoff = (unsigned)(size_t)(sxd + i);   // LDS byte offset (low 32b)
    const float* g = src + i;
    asm volatile("global_load_async_to_lds_b128 %0, %1, off"
                 :: "v"(ldsoff), "v"(g) : "memory");
  }
  asm volatile("s_wait_asynccnt 0x0" ::: "memory");
  __syncthreads();

  float dw[Nst], Arow[Nst], h[Nst];
#pragma unroll
  for (int n = 0; n < Nst; ++n) {
    dw[n]   = dtw[((size_t)k * Din + d) * Nst + n];
    Arow[n] = -__expf(alogs[((size_t)k * Din + d) * Nst + n]);
    h[n] = 0.f;
  }
  const float db = dtb[k * Din + d];
  const float Dd = dsc[k * Din + d];
  const float* xcrow = xc + ((size_t)b * Din + d) * Ll;
  float* yrow        = ys + (((size_t)b * K4 + k) * Din + d) * Ll;

  for (int l = 0; l < Ll; ++l) {
    float dtv = db;
#pragma unroll
    for (int r = 0; r < Nst; ++r) dtv += sxd[r * Ll + l] * dw[r];
    float delta = (dtv > 20.f) ? dtv : log1pf(__expf(dtv));
    int li;
    switch (k) {
      case 0:  li = l; break;
      case 1:  li = trmap(l); break;
      case 2:  li = Ll - 1 - l; break;
      default: li = trmap(Ll - 1 - l); break;
    }
    float u  = xcrow[li];
    float du = delta * u;
    float y  = 0.f;
#pragma unroll
    for (int n = 0; n < Nst; ++n) {
      float dA = __expf(delta * Arow[n]);
      h[n] = h[n] * dA + du * sxd[(Nst + n) * Ll + l];
      y += h[n] * sxd[(2 * Nst + n) * Ll + l];
    }
    yrow[l] = y + Dd * u;
  }
}

// ---------------------------------------------------------------------------
// Merge 4 scan directions, LayerNorm over 512, gate with silu(z) -> bf16 rows
// ---------------------------------------------------------------------------
__global__ __launch_bounds__(256)
void combine_gate_kernel(const float* __restrict__ ys, const float* __restrict__ xz,
                         const float* __restrict__ og, const float* __restrict__ ob,
                         __bf16* __restrict__ out) {
  __shared__ float red[256];
  const int row = blockIdx.x, t = threadIdx.x;
  const int b = row / Ll, l = row % Ll;
  const int lm = trmap(l);
  float vals[2];
  float sum = 0.f;
#pragma unroll
  for (int i = 0; i < 2; ++i) {
    int d = t + i * 256;
    size_t bd = (size_t)b * K4;
    float y = ys[((bd + 0) * Din + d) * Ll + l]
            + ys[((bd + 2) * Din + d) * Ll + (Ll - 1 - l)]
            + ys[((bd + 1) * Din + d) * Ll + lm]
            + ys[((bd + 3) * Din + d) * Ll + (Ll - 1 - lm)];
    vals[i] = y; sum += y;
  }
  red[t] = sum; __syncthreads();
  for (int s = 128; s > 0; s >>= 1) { if (t < s) red[t] += red[t + s]; __syncthreads(); }
  float mu = red[0] * (1.0f / Din); __syncthreads();
  float vs = 0.f;
#pragma unroll
  for (int i = 0; i < 2; ++i) { float dd = vals[i] - mu; vs += dd * dd; }
  red[t] = vs; __syncthreads();
  for (int s = 128; s > 0; s >>= 1) { if (t < s) red[t] += red[t + s]; __syncthreads(); }
  float rs = rsqrtf(red[0] * (1.0f / Din) + EPSv);
#pragma unroll
  for (int i = 0; i < 2; ++i) {
    int d = t + i * 256;
    float z  = xz[(size_t)row * (2 * Din) + Din + d];
    float zn = z * sigm(z);
    out[(size_t)row * Din + d] = f2bf(((vals[i] - mu) * rs * og[d] + ob[d]) * zn);
  }
}

// ---------------------------------------------------------------------------
// residual add + pack bf16 into concat buffer half (content=0 / style=1)
// ---------------------------------------------------------------------------
__global__ __launch_bounds__(256)
void residual_concat_kernel(const float* __restrict__ oproj, const float* __restrict__ xin,
                            __bf16* __restrict__ cat, int half) {
  int idx = blockIdx.x * 256 + threadIdx.x;   // over Mr*Cc
  int ch = idx & (Cc - 1), row = idx >> 8;
  float v = xin[idx] + oproj[idx];
  cat[(size_t)row * (2 * Cc) + half * Cc + ch] = f2bf(v);
}

// ---------------------------------------------------------------------------
// + combine_b, LayerNorm over 256 -> fp32 output
// ---------------------------------------------------------------------------
__global__ __launch_bounds__(256)
void final_ln_kernel(const float* __restrict__ x, const float* __restrict__ cbias,
                     const float* __restrict__ g, const float* __restrict__ b,
                     float* __restrict__ out) {
  __shared__ float red[256];
  const int row = blockIdx.x, t = threadIdx.x;
  float v = x[(size_t)row * Cc + t] + cbias[t];
  red[t] = v; __syncthreads();
  for (int s = 128; s > 0; s >>= 1) { if (t < s) red[t] += red[t + s]; __syncthreads(); }
  float mu = red[0] * (1.0f / Cc); __syncthreads();
  float d = v - mu;
  red[t] = d * d; __syncthreads();
  for (int s = 128; s > 0; s >>= 1) { if (t < s) red[t] += red[t + s]; __syncthreads(); }
  float rs = rsqrtf(red[0] * (1.0f / Cc) + EPSv);
  out[(size_t)row * Cc + t] = d * rs * g[t] + b[t];
}

// ---------------------------------------------------------------------------
// host launcher
// ---------------------------------------------------------------------------
extern "C" void kernel_launch(void* const* d_in, const int* in_sizes, int n_in,
                              void* d_out, int out_size, void* d_ws, size_t ws_size,
                              hipStream_t stream) {
  (void)in_sizes; (void)n_in; (void)out_size; (void)ws_size;
  const float* xin[2] = { (const float*)d_in[0], (const float*)d_in[1] };
  // params flattened in dict order: per-block 13 tensors
  //   0 ln1_g 1 ln1_b 2 in_proj_w 3 conv_w 4 conv_b 5 x_proj_w 6 dt_w 7 dt_b
  //   8 A_logs 9 Ds 10 onorm_g 11 onorm_b 12 out_proj_w
  const int PB[2] = { 2, 15 };
  const float* combine_w = (const float*)d_in[28];
  const float* combine_b = (const float*)d_in[29];
  const float* norm_g    = (const float*)d_in[30];
  const float* norm_b    = (const float*)d_in[31];

  // workspace carve (~78 MB total)
  char* cur = (char*)d_ws;
  auto carve = [&](size_t bytes) {
    char* p = cur; cur += (bytes + 255) & ~(size_t)255; return (void*)p;
  };
  __bf16* wInBf   = (__bf16*)carve((size_t)2 * Din * Cc * sizeof(__bf16));  // 1024x256
  __bf16* wOutBf  = (__bf16*)carve((size_t)Cc * Din * sizeof(__bf16));      // 256x512
  __bf16* wCmbBf  = (__bf16*)carve((size_t)Cc * 2 * Cc * sizeof(__bf16));   // 256x512
  __bf16* lnA     = (__bf16*)carve((size_t)Mr * Cc * sizeof(__bf16));
  float*  xz      = (float*) carve((size_t)Mr * 2 * Din * sizeof(float));
  float*  xc      = (float*) carve((size_t)Bb * Din * Ll * sizeof(float));
  float*  xdbl    = (float*) carve((size_t)Bb * K4 * Cdbl * Ll * sizeof(float));
  float*  ys      = (float*) carve((size_t)Bb * K4 * Din * Ll * sizeof(float));
  __bf16* gateA   = (__bf16*)carve((size_t)Mr * Din * sizeof(__bf16));
  float*  oproj   = (float*) carve((size_t)Mr * Cc * sizeof(float));
  __bf16* catA    = (__bf16*)carve((size_t)Mr * 2 * Cc * sizeof(__bf16));
  float*  comb    = (float*) carve((size_t)Mr * Cc * sizeof(float));

  const size_t scanLds = (size_t)Cdbl * Ll * sizeof(float);   // 192 KB / WGP

  for (int bi = 0; bi < 2; ++bi) {
    auto prm = [&](int j) { return (const float*)d_in[PB[bi] + j]; };
    const int nInW = 2 * Din * Cc, nOutW = Cc * Din;
    cast_bf16_kernel<<<(nInW  + 255) / 256, 256, 0, stream>>>(prm(2),  wInBf,  nInW);
    cast_bf16_kernel<<<(nOutW + 255) / 256, 256, 0, stream>>>(prm(12), wOutBf, nOutW);

    ln256_to_bf16_kernel<<<Mr, 256, 0, stream>>>(xin[bi], prm(0), prm(1), lnA);
    wmma_gemm_kernel<<<dim3((2 * Din) / 64, Mr / 128), 256, 0, stream>>>(
        lnA, wInBf, xz, Mr, 2 * Din, Cc);

    conv_silu_kernel<<<(Bb * Din * Ll) / 256, 256, 0, stream>>>(xz, prm(3), prm(4), xc);
    xproj_kernel<<<(Bb * K4 * Cdbl * Ll) / 256, 256, 0, stream>>>(xc, prm(5), xdbl);
    scan_kernel<<<Bb * K4, 512, scanLds, stream>>>(
        xc, xdbl, prm(6), prm(7), prm(8), prm(9), ys);
    combine_gate_kernel<<<Mr, 256, 0, stream>>>(ys, xz, prm(10), prm(11), gateA);

    wmma_gemm_kernel<<<dim3(Cc / 64, Mr / 128), 256, 0, stream>>>(
        gateA, wOutBf, oproj, Mr, Cc, Din);
    residual_concat_kernel<<<(Mr * Cc) / 256, 256, 0, stream>>>(oproj, xin[bi], catA, bi);
  }

  const int nCmbW = Cc * 2 * Cc;
  cast_bf16_kernel<<<(nCmbW + 255) / 256, 256, 0, stream>>>(combine_w, wCmbBf, nCmbW);
  wmma_gemm_kernel<<<dim3(Cc / 64, Mr / 128), 256, 0, stream>>>(
      catA, wCmbBf, comb, Mr, Cc, 2 * Cc);
  final_ln_kernel<<<Mr, 256, 0, stream>>>(comb, combine_b, norm_g, norm_b, (float*)d_out);
}
